// CheekGPT_2740189135695
// MI455X (gfx1250) — matrix operand order repaired and checked by reading, not compiled, not run on hardware
//
#include <hip/hip_runtime.h>
#include <hip/hip_fp16.h>

typedef __attribute__((ext_vector_type(16))) _Float16 v16h;
typedef __attribute__((ext_vector_type(8)))  float    v8f;

// ---------------------------------------------------------------------------
// Tensor Data Mover (TDM) support, with graceful fallback.
// ---------------------------------------------------------------------------
#if defined(__HIP_DEVICE_COMPILE__) && defined(__gfx1250__) && \
    __has_builtin(__builtin_amdgcn_tensor_load_to_lds)
#define GPT_HAS_TDM 1
#else
#define GPT_HAS_TDM 0
#endif

#if __has_include(<hip/amd_detail/amd_gfx1250_TDM.h>)
#define GPT_TDM_6ARG 1
#else
#define GPT_TDM_6ARG 0
#endif

#if GPT_HAS_TDM
typedef unsigned tdm_u4 __attribute__((ext_vector_type(4)));
typedef int      tdm_i4 __attribute__((ext_vector_type(4)));
typedef int      tdm_i8 __attribute__((ext_vector_type(8)));
typedef __attribute__((address_space(3))) _Float16 lds_f16_t;

static __device__ __forceinline__ unsigned lds_offset_of(const _Float16* p) {
  return (unsigned)(unsigned long long)(lds_f16_t*)p;
}

// DMA a 2D f16 tile (rows x 32 elements, row pitch `pitch` elements) from
// global memory into LDS at byte offset lds_off (row-major, contiguous).
// D# per CDNA5 ISA ch.8: group0 {count,lds,global,type=2}, group1 {dims}.
static __device__ __forceinline__ void tdm_load_2d_f16(
    const _Float16* g, unsigned lds_off, int rows, int pitch) {
  const unsigned long long ga = (unsigned long long)(size_t)g;
  tdm_u4 g0 = {};
  g0.x = 1u;                                            // count=1 (valid)
  g0.y = lds_off;                                       // lds_addr [63:32]
  g0.z = (unsigned)ga;                                  // global_addr lo
  g0.w = (unsigned)((ga >> 32) & 0x01ffffffull) | (2u << 30);  // hi | type=2
  const unsigned long long s0 = (unsigned long long)pitch;
  tdm_i8 g1 = {};
  g1[0] = 0x00010000;                                   // data_size = 2 bytes
  g1[1] = (int)(32u << 16);                             // tensor_dim0 lo16 = 32
  g1[2] = (int)(((unsigned)rows & 0xffffu) << 16);      // tensor_dim1 lo16
  g1[3] = (int)(32u << 16);                             // tile_dim0 = 32
  g1[4] = (int)((unsigned)rows & 0xffffu);              // tile_dim1 = rows
  g1[5] = (int)(unsigned)(s0 & 0xffffffffull);          // dim0_stride lo32
  g1[6] = (int)(unsigned)((s0 >> 32) & 0xffffull);      // dim0_stride hi16
  g1[7] = 0;
  tdm_i4 z4 = {};
#if GPT_TDM_6ARG
  tdm_i8 z8 = {};
  __builtin_amdgcn_tensor_load_to_lds(g0, g1, z4, z4, z8, 0);
#else
  __builtin_amdgcn_tensor_load_to_lds(g0, g1, z4, z4, 0);
#endif
}

static __device__ __forceinline__ void tdm_wait0() {
#if __has_builtin(__builtin_amdgcn_s_wait_tensorcnt)
  __builtin_amdgcn_s_wait_tensorcnt(0);
#else
  asm volatile("s_wait_tensorcnt 0x0" ::: "memory");
#endif
}
#endif  // GPT_HAS_TDM

// ---------------------------------------------------------------------------
// WMMA helpers (CDNA5 gfx1250, wave32)
// ---------------------------------------------------------------------------
static __device__ __forceinline__ v8f wmma16x16x32(v16h a, v16h b, v8f c) {
  return __builtin_amdgcn_wmma_f32_16x16x32_f16(
      /*neg_a=*/false, a, /*neg_b=*/false, b,
      /*c_mod=*/(short)0, c, /*reuse_a=*/false, /*reuse_b=*/false);
}

// A fragment, 16x32 f16 (ISA 7.12.2):
// lane L: m=L&15, khalf=L>>4 ; VGPR v: kk = (v>>2)*16 + khalf*8 + (v&3)*2
static __device__ __forceinline__ v16h load_frag_A(const _Float16* lds,
                                                   int row_base, int stride) {
  const int lane = threadIdx.x & 31;
  const int m = lane & 15, kh = lane >> 4;
  union { v16h v; unsigned u[8]; } f;
#pragma unroll
  for (int r = 0; r < 8; ++r) {
    const int kk = (r >> 2) * 16 + kh * 8 + (r & 3) * 2;
    f.u[r] = *(const unsigned*)(lds + (size_t)(row_base + m) * stride + kk);
  }
  return f.v;
}

// B fragment, 32x16 f16: lane L: n=L&15, khalf=L>>4 ; VGPR v: k = (v>>2)*16 + khalf*8 + (v&3)*2
// ldsT holds B^T ([N][K]); K-pairs contiguous.
static __device__ __forceinline__ v16h load_frag_B(const _Float16* ldsT,
                                                   int col_base, int stride) {
  const int lane = threadIdx.x & 31;
  const int n = lane & 15, kh = lane >> 4;
  union { v16h v; unsigned u[8]; } f;
#pragma unroll
  for (int r = 0; r < 8; ++r) {
    const int kk = (r >> 2) * 16 + kh * 8 + (r & 3) * 2;
    f.u[r] = *(const unsigned*)(ldsT + (size_t)(col_base + n) * stride + kk);
  }
  return f.v;
}

// ---------------------------------------------------------------------------
// Tiled WMMA GEMM: out[M,N] = A16[M,K] @ W16[K,N] + bias (+ReLU) (+f16 copy)
// block = 128 threads (4 waves), tile 64x64, K-step 32.
// A tile staged via TDM tensor_load_to_lds (async DMA, TENSORcnt).
// ---------------------------------------------------------------------------
template <bool RELU, bool DUAL>
__global__ __launch_bounds__(128) void gemm_wmma(
    const _Float16* __restrict__ A, const _Float16* __restrict__ W,
    const float* __restrict__ bias, float* __restrict__ out,
    _Float16* __restrict__ out16, int M, int K, int N) {
  __shared__ _Float16 As[64 * 32];  // [m][k]
  __shared__ _Float16 Bs[64 * 32];  // [n][k]  (W tile transposed)
  const int row0 = blockIdx.y * 64, col0 = blockIdx.x * 64;
  const int tid = threadIdx.x, lane = tid & 31, wv = tid >> 5;
  const int wrow = (wv & 1) * 32, wcol = (wv >> 1) * 32;
#if GPT_HAS_TDM
  const unsigned As_lds = lds_offset_of(As);
#endif
  v8f acc[2][2] = {};
  for (int k0 = 0; k0 < K; k0 += 32) {
    __syncthreads();
#if GPT_HAS_TDM
    if (wv == 0)  // one wave issues the DMA descriptor for the A tile
      tdm_load_2d_f16(A + (size_t)row0 * K + k0, As_lds, 64, K);
#else
#pragma unroll
    for (int i = 0; i < 8; ++i) {  // A: 1024 dwords / 128 threads
      const int t = tid + i * 128;
      const int m = t >> 4, k = (t & 15) * 2;
      *(unsigned*)(As + m * 32 + k) =
          *(const unsigned*)(A + (size_t)(row0 + m) * K + (k0 + k));
    }
#endif
#pragma unroll
    for (int i = 0; i < 8; ++i) {  // W: read along N, store transposed
      const int t = tid + i * 128;
      const int k = t >> 5, n2 = (t & 31) * 2;
      union { unsigned u; _Float16 h[2]; } d;
      d.u = *(const unsigned*)(W + (size_t)(k0 + k) * N + (col0 + n2));
      Bs[(n2 + 0) * 32 + k] = d.h[0];
      Bs[(n2 + 1) * 32 + k] = d.h[1];
    }
    if (k0 + 32 < K)  // warm L2 for next W tile (global_prefetch_b8)
      __builtin_prefetch(W + (size_t)(k0 + 32 + (tid >> 2)) * N + col0 +
                             (tid & 3) * 16, 0, 1);
#if GPT_HAS_TDM
    if (wv == 0) tdm_wait0();  // TENSORcnt==0 -> A tile landed in LDS
#endif
    __syncthreads();
    const v16h a0 = load_frag_A(As, wrow + 0, 32);
    const v16h a1 = load_frag_A(As, wrow + 16, 32);
    const v16h b0 = load_frag_B(Bs, wcol + 0, 32);
    const v16h b1 = load_frag_B(Bs, wcol + 16, 32);
    acc[0][0] = wmma16x16x32(a0, b0, acc[0][0]);
    acc[0][1] = wmma16x16x32(a0, b1, acc[0][1]);
    acc[1][0] = wmma16x16x32(a1, b0, acc[1][0]);
    acc[1][1] = wmma16x16x32(a1, b1, acc[1][1]);
  }
  const int nc = lane & 15, kh = lane >> 4;
#pragma unroll
  for (int i = 0; i < 2; ++i)
#pragma unroll
    for (int j = 0; j < 2; ++j) {
      const int c = col0 + wcol + j * 16 + nc;
      const float bv = bias ? bias[c] : 0.f;
#pragma unroll
      for (int r = 0; r < 8; ++r) {
        const int rr = row0 + wrow + i * 16 + r + 8 * kh;
        float v = acc[i][j][r] + bv;
        if (RELU) v = fmaxf(v, 0.f);
        out[(size_t)rr * N + c] = v;
        if (DUAL) out16[(size_t)rr * N + c] = (_Float16)v;
      }
    }
}

// ---------------------------------------------------------------------------
// Flash attention, causal, hd=64. qkv rows = [k | q | v] (k first!).
// block = 128 threads (4 waves) = 64 query rows for one (batch, head).
// ---------------------------------------------------------------------------
__global__ __launch_bounds__(128) void flash_attn(
    const float* __restrict__ qkv, float* __restrict__ att, int T, int D, int H) {
  __shared__ _Float16 Qs[64 * 64];      // [m][d], pre-scaled by 1/8
  __shared__ _Float16 Ks[32 * 64];      // [key][d]   (B^T for QK^T)
  __shared__ _Float16 Vs[64 * 32];      // [d][key]   (B^T for P@V)
  __shared__ _Float16 Ps[4 * 16 * 32];  // per-wave P relayout buffer
  const int tid = threadIdx.x, lane = tid & 31, wv = tid >> 5;
  const int b = blockIdx.z, h = blockIdx.y, qblk = blockIdx.x * 64;
  const int C3 = 3 * D;
  const size_t rb = (size_t)b * T;
  const int koff = h * 64, qoff = D + h * 64, voff = 2 * D + h * 64;

  for (int i = tid; i < 64 * 64; i += 128) {
    const int m = i >> 6, d = i & 63;
    Qs[i] = (_Float16)(qkv[(rb + qblk + m) * C3 + qoff + d] * 0.125f);
  }
  __syncthreads();

  const int q0 = qblk + wv * 16;
  const v16h aq0 = load_frag_A(Qs, wv * 16, 64);       // hd 0..31
  const v16h aq1 = load_frag_A(Qs + 32, wv * 16, 64);  // hd 32..63
  v8f o0 = {}, o1 = {}, o2 = {}, o3 = {};
  float mrow[8], lrow[8];
#pragma unroll
  for (int r = 0; r < 8; ++r) { mrow[r] = -1e30f; lrow[r] = 0.f; }
  const int nc = lane & 15, kh = lane >> 4;
  _Float16* P = Ps + wv * (16 * 32);

  for (int kb = 0; kb < qblk + 64; kb += 32) {
    __syncthreads();
    for (int i = tid; i < 32 * 64; i += 128) {
      const int kk = i >> 6, d = i & 63;
      Ks[kk * 64 + d] = (_Float16)qkv[(rb + kb + kk) * C3 + koff + d];
      Vs[d * 32 + kk] = (_Float16)qkv[(rb + kb + kk) * C3 + voff + d];
    }
    __syncthreads();
    const v8f z = {};
    v8f s0 = wmma16x16x32(aq1, load_frag_B(Ks + 32, 0, 64),
             wmma16x16x32(aq0, load_frag_B(Ks, 0, 64), z));
    v8f s1 = wmma16x16x32(aq1, load_frag_B(Ks + 32, 16, 64),
             wmma16x16x32(aq0, load_frag_B(Ks, 16, 64), z));
#pragma unroll
    for (int r = 0; r < 8; ++r) {
      const int qg = q0 + r + 8 * kh;
      const int kg0 = kb + nc, kg1 = kb + 16 + nc;
      const float v0 = (kg0 <= qg) ? s0[r] : -1e30f;
      const float v1 = (kg1 <= qg) ? s1[r] : -1e30f;
      float mx = fmaxf(v0, v1);
#pragma unroll
      for (int offm = 1; offm < 16; offm <<= 1)
        mx = fmaxf(mx, __shfl_xor(mx, offm, 32));
      const float mnew = fmaxf(mrow[r], mx);
      const float a = __expf(mrow[r] - mnew);
      const float p0 = (kg0 <= qg) ? __expf(v0 - mnew) : 0.f;
      const float p1 = (kg1 <= qg) ? __expf(v1 - mnew) : 0.f;
      float ps = p0 + p1;
#pragma unroll
      for (int offs = 1; offs < 16; offs <<= 1)
        ps += __shfl_xor(ps, offs, 32);
      lrow[r] = lrow[r] * a + ps;
      mrow[r] = mnew;
      P[(r + 8 * kh) * 32 + nc] = (_Float16)p0;        // C-layout -> A-layout
      P[(r + 8 * kh) * 32 + nc + 16] = (_Float16)p1;   // via per-wave LDS
      o0[r] *= a; o1[r] *= a; o2[r] *= a; o3[r] *= a;
    }
    const v16h pf = load_frag_A(P, 0, 32);  // 16x32 keys
    o0 = wmma16x16x32(pf, load_frag_B(Vs, 0, 32), o0);
    o1 = wmma16x16x32(pf, load_frag_B(Vs, 16, 32), o1);
    o2 = wmma16x16x32(pf, load_frag_B(Vs, 32, 32), o2);
    o3 = wmma16x16x32(pf, load_frag_B(Vs, 48, 32), o3);
  }
#pragma unroll
  for (int r = 0; r < 8; ++r) {
    const int qg = q0 + r + 8 * kh;
    const float li = 1.f / lrow[r];
    float* orow = att + (rb + qg) * D + h * 64;
    orow[0 + nc]  = o0[r] * li;
    orow[16 + nc] = o1[r] * li;
    orow[32 + nc] = o2[r] * li;
    orow[48 + nc] = o3[r] * li;
  }
}

// ---------------------------------------------------------------------------
// Small elementwise / reduction kernels
// ---------------------------------------------------------------------------
__global__ void embed_kernel(const int* __restrict__ idx,
                             const float* __restrict__ tok,
                             const float* __restrict__ pos,
                             float* __restrict__ x, int T, int D) {
  const int row = blockIdx.x;
  const int t = row % T;
  const int id = idx[row];
  for (int d = threadIdx.x; d < D; d += blockDim.x)
    x[(size_t)row * D + d] = tok[(size_t)id * D + d] + pos[(size_t)t * D + d];
}

__global__ void f32_to_f16_kernel(const float* __restrict__ in,
                                  _Float16* __restrict__ out, size_t n) {
  size_t i = (size_t)blockIdx.x * blockDim.x + threadIdx.x;
  const size_t st = (size_t)gridDim.x * blockDim.x;
  for (; i < n; i += st) out[i] = (_Float16)in[i];
}

__global__ void add_inplace(float* __restrict__ x, const float* __restrict__ y,
                            size_t n) {
  size_t i = (size_t)blockIdx.x * blockDim.x + threadIdx.x;
  const size_t st = (size_t)gridDim.x * blockDim.x;
  for (; i < n; i += st) x[i] += y[i];
}

__global__ __launch_bounds__(256) void layernorm_f16(
    const float* __restrict__ x, const float* __restrict__ g,
    const float* __restrict__ bta, _Float16* __restrict__ out, int D) {
  __shared__ float red[256];
  const int row = blockIdx.x, tid = threadIdx.x;
  const float* xr = x + (size_t)row * D;
  float s = 0.f;
  for (int i = tid; i < D; i += 256) s += xr[i];
  red[tid] = s; __syncthreads();
  for (int o = 128; o > 0; o >>= 1) {
    if (tid < o) red[tid] += red[tid + o];
    __syncthreads();
  }
  const float mu = red[0] / D;
  __syncthreads();
  float vs = 0.f;
  for (int i = tid; i < D; i += 256) { const float d = xr[i] - mu; vs += d * d; }
  red[tid] = vs; __syncthreads();
  for (int o = 128; o > 0; o >>= 1) {
    if (tid < o) red[tid] += red[tid + o];
    __syncthreads();
  }
  const float rstd = rsqrtf(red[0] / D + 1e-5f);
  _Float16* orow = out + (size_t)row * D;
  for (int i = tid; i < D; i += 256)
    orow[i] = (_Float16)((xr[i] - mu) * rstd * g[i] + bta[i]);
}

__global__ __launch_bounds__(256) void nll_kernel(
    const float* __restrict__ logits, const int* __restrict__ target,
    float* __restrict__ nll, int V) {
  __shared__ float red[256];
  const int row = blockIdx.x, tid = threadIdx.x;
  const float* lr = logits + (size_t)row * V;
  float mx = -1e30f;
  for (int i = tid; i < V; i += 256) mx = fmaxf(mx, lr[i]);
  red[tid] = mx; __syncthreads();
  for (int o = 128; o > 0; o >>= 1) {
    if (tid < o) red[tid] = fmaxf(red[tid], red[tid + o]);
    __syncthreads();
  }
  const float m = red[0];
  __syncthreads();
  float s = 0.f;
  for (int i = tid; i < V; i += 256) s += __expf(lr[i] - m);
  red[tid] = s; __syncthreads();
  for (int o = 128; o > 0; o >>= 1) {
    if (tid < o) red[tid] += red[tid + o];
    __syncthreads();
  }
  if (tid == 0) nll[row] = logf(red[0]) + m - lr[target[row]];
}

__global__ __launch_bounds__(256) void loss_kernel(const float* __restrict__ nll,
                                                   float* __restrict__ out, int n) {
  __shared__ float red[256];
  float s = 0.f;
  for (int i = threadIdx.x; i < n; i += 256) s += nll[i];
  red[threadIdx.x] = s; __syncthreads();
  for (int o = 128; o > 0; o >>= 1) {
    if (threadIdx.x < o) red[threadIdx.x] += red[threadIdx.x + o];
    __syncthreads();
  }
  if (threadIdx.x == 0) *out = red[0] / n;
}

// ---------------------------------------------------------------------------
// Orchestration
// ---------------------------------------------------------------------------
extern "C" void kernel_launch(void* const* d_in, const int* in_sizes, int n_in,
                              void* d_out, int out_size, void* d_ws, size_t ws_size,
                              hipStream_t stream) {
  (void)in_sizes; (void)n_in; (void)out_size; (void)ws_size;
  constexpr int B = 2, T = 2048, D = 768, H = 12, L = 4, V = 32000;
  constexpr int R = B * T;  // 4096 rows

  const int* idx     = (const int*)d_in[0];
  const int* target  = (const int*)d_in[1];
  const float* tok   = (const float*)d_in[2];
  const float* pos   = (const float*)d_in[3];
  const float* lnf_g = (const float*)d_in[4];
  const float* lnf_b = (const float*)d_in[5];
  const float* head_w = (const float*)d_in[6];
  const float* head_b = (const float*)d_in[7];
  auto bp = [&](int l, int j) { return (const float*)d_in[8 + l * 12 + j]; };
  // j: 0 qkv_w 1 qkv_b 2 proj_w 3 proj_b 4 fc1_w 5 fc1_b 6 fc2_w 7 fc2_b
  //    8 ln1_g 9 ln1_b 10 ln2_g 11 ln2_b

  char* ws = (char*)d_ws;
  size_t off = 0;
  auto carve = [&](size_t bytes) -> char* {
    char* p = ws + off;
    off = (off + bytes + 255) & ~(size_t)255;
    return p;
  };
  float*    X    = (float*)carve((size_t)R * D * 4);
  _Float16* A16  = (_Float16*)carve((size_t)R * D * 2);
  _Float16* H16  = (_Float16*)carve((size_t)R * 4 * D * 2);
  float*    QKV  = (float*)carve((size_t)R * 3 * D * 4);
  float*    ATT  = (float*)carve((size_t)R * D * 4);
  float*    GOUT = (float*)carve((size_t)R * 4 * D * 4);
  _Float16* W16  = (_Float16*)carve((size_t)D * V * 2);
  float*    NLL  = (float*)carve((size_t)R * 4);

  float* logits = (float*)d_out;
  float* loss   = logits + (size_t)R * V;

  embed_kernel<<<R, 256, 0, stream>>>(idx, tok, pos, X, T, D);

  for (int l = 0; l < L; ++l) {
    // attn: x += proj(attn(ln1(x)))
    layernorm_f16<<<R, 256, 0, stream>>>(X, bp(l, 8), bp(l, 9), A16, D);
    f32_to_f16_kernel<<<1024, 256, 0, stream>>>(bp(l, 0), W16, (size_t)D * 3 * D);
    gemm_wmma<false, false><<<dim3(3 * D / 64, R / 64), 128, 0, stream>>>(
        A16, W16, bp(l, 1), QKV, nullptr, R, D, 3 * D);
    flash_attn<<<dim3(T / 64, H, B), 128, 0, stream>>>(QKV, ATT, T, D, H);
    f32_to_f16_kernel<<<1024, 256, 0, stream>>>(ATT, A16, (size_t)R * D);
    f32_to_f16_kernel<<<1024, 256, 0, stream>>>(bp(l, 2), W16, (size_t)D * D);
    gemm_wmma<false, false><<<dim3(D / 64, R / 64), 128, 0, stream>>>(
        A16, W16, bp(l, 3), GOUT, nullptr, R, D, D);
    add_inplace<<<1024, 256, 0, stream>>>(X, GOUT, (size_t)R * D);
    // mlp: x += fc2(relu(fc1(ln2(x))))
    layernorm_f16<<<R, 256, 0, stream>>>(X, bp(l, 10), bp(l, 11), A16, D);
    f32_to_f16_kernel<<<1024, 256, 0, stream>>>(bp(l, 4), W16, (size_t)D * 4 * D);
    gemm_wmma<true, true><<<dim3(4 * D / 64, R / 64), 128, 0, stream>>>(
        A16, W16, bp(l, 5), GOUT, H16, R, D, 4 * D);
    f32_to_f16_kernel<<<1024, 256, 0, stream>>>(bp(l, 6), W16, (size_t)(4 * D) * D);
    gemm_wmma<false, false><<<dim3(D / 64, R / 64), 128, 0, stream>>>(
        H16, W16, bp(l, 7), GOUT, nullptr, R, 4 * D, D);
    add_inplace<<<1024, 256, 0, stream>>>(X, GOUT, (size_t)R * D);
  }

  layernorm_f16<<<R, 256, 0, stream>>>(X, lnf_g, lnf_b, A16, D);
  f32_to_f16_kernel<<<2048, 256, 0, stream>>>(head_w, W16, (size_t)D * V);
  gemm_wmma<false, false><<<dim3(V / 64, R / 64), 128, 0, stream>>>(
      A16, W16, head_b, logits, nullptr, R, D, V);

  nll_kernel<<<R, 256, 0, stream>>>(logits, target, NLL, V);
  loss_kernel<<<1, 256, 0, stream>>>(NLL, loss, R);
}
